// SimpleResRNN_14121852470210
// MI455X (gfx1250) — compile-verified, matrix operand order checked by reading, not run on hardware
//
#include <hip/hip_runtime.h>

// BRF resonate-and-fire RNN scan for MI455X (gfx1250).
// Persistent single-workgroup kernel: 16 wave32s on one WGP.
// CDNA5 320KB LDS holds ALL weights + activations in exact WMMA operand layout:
//   WhB[2][16][9][32] uint4 : W_h bf16 B-panels (per wave / k-step), conflict-free b128 loads
//   Aop[2][4][9][32]  uint4 : A = [x_t | z_{t-1}] bf16 in A-operand layout
//   WoB[2][8][32]     uint4 : W_o bf16 B-panels for the distributed LI output GEMM
// u/v/q state lives in registers in WMMA C/D layout. No scratch spills by design.

#define TT    2048
#define BB    64
#define II    32
#define HH    256
#define OO    16
#define KTOT  288          // II + HH
#define NTH   512          // 16 waves * 32
#define DT_C    0.01f
#define GAMMA_C 0.9f

typedef __attribute__((ext_vector_type(16))) __bf16       v16bf;
typedef __attribute__((ext_vector_type(8)))  float        v8f;
typedef __attribute__((ext_vector_type(4)))  float        v4f;
typedef __attribute__((ext_vector_type(2)))  unsigned int v2u;

union ABu { v16bf v; v4f f4[2]; unsigned short us[16]; };

__device__ __forceinline__ unsigned short f2bf(float f) {
  unsigned int u = __builtin_bit_cast(unsigned int, f);
  return (unsigned short)((u + 0x7FFFu + ((u >> 16) & 1u)) >> 16);   // RNE
}
__device__ __forceinline__ float bf2f(unsigned short b) {
  unsigned int u = (unsigned int)b << 16;
  return __builtin_bit_cast(float, u);
}

__global__ __launch_bounds__(NTH, 1)
void brf_scan_kernel(const float* __restrict__ x,
                     const float* __restrict__ Wh,
                     const float* __restrict__ omega,
                     const float* __restrict__ boff,
                     const float* __restrict__ Wo,
                     const float* __restrict__ tau,
                     float* __restrict__ out)
{
  __shared__ uint4 WhB[2][16][9][32];   // 147456 B : W_h bf16 B-operands
  __shared__ uint4 Aop[2][4][9][32];    //  36864 B : A operands (4 M-tiles x 9 k-steps)
  __shared__ uint4 WoB[2][8][32];       //   8192 B : W_o bf16 B-operands
  __shared__ float outpart[BB * OO];    //   4096 B : z@Wo^T partial reduction
  __shared__ float oubuf[BB * OO];      //   4096 B : LI membrane out_u
  __shared__ float ns_slot;

  unsigned short* Aus = (unsigned short*)Aop;

  const int tid  = threadIdx.x;
  const int wave = tid >> 5;
  const int lane = tid & 31;
  const int lo   = lane & 15;
  const int hi   = lane >> 4;
  const int kkO  = wave >> 1;           // this wave's output k-block (0..7)
  const int mA   = (wave & 1) * 2;      // first of its two output M-tiles

  if (tid == 0) ns_slot = 0.0f;

  // zero all A operands (z_{-1} = 0; x region overwritten below)
  {
    uint4 zz; zz.x = zz.y = zz.z = zz.w = 0u;
    uint4* af = (uint4*)Aop;
    for (int idx = tid; idx < 2 * 4 * 9 * 32; idx += NTH) af[idx] = zz;
  }
  for (int idx = tid; idx < BB * OO; idx += NTH) { outpart[idx] = 0.0f; oubuf[idx] = 0.0f; }

  // ---- stage W_h into LDS B-operand layout (each wave stages its own panel) ----
  {
    const float* wrow = Wh + (size_t)(16 * wave + lo) * KTOT;
    for (int kk = 0; kk < 9; ++kk) {
      ABu p;
      const float* pp = wrow + 32 * kk + 8 * hi;
      #pragma unroll
      for (int j = 0; j < 8; ++j) {
        p.us[j]     = f2bf(pp[j]);
        p.us[8 + j] = f2bf(pp[16 + j]);
      }
      *(v4f*)&WhB[0][wave][kk][lane] = p.f4[0];
      *(v4f*)&WhB[1][wave][kk][lane] = p.f4[1];
    }
  }
  // ---- stage W_o (waves 0..7, one k-block each) ----
  if (wave < 8) {
    ABu p;
    const float* pp = Wo + (size_t)lo * HH + 32 * wave + 8 * hi;
    #pragma unroll
    for (int j = 0; j < 8; ++j) {
      p.us[j]     = f2bf(pp[j]);
      p.us[8 + j] = f2bf(pp[16 + j]);
    }
    *(v4f*)&WoB[0][wave][lane] = p.f4[0];
    *(v4f*)&WoB[1][wave][lane] = p.f4[1];
  }

  // ---- stage x_0 into A-operand layout: thread -> 4 consecutive k of one row ----
  {
    const int flat = tid * 4;
    const int b = flat >> 5, kb = flat & 31;
    v4f xv = *(const v4f*)(x + flat);
    const int m = b >> 4, half = (kb >> 4) & 1, s = (kb >> 3) & 1, el = kb & 7;
    const int lp = (b & 15) + 16 * s;
    unsigned int p0 = (unsigned int)f2bf(xv.x) | ((unsigned int)f2bf(xv.y) << 16);
    unsigned int p1 = (unsigned int)f2bf(xv.z) | ((unsigned int)f2bf(xv.w) << 16);
    v2u pk; pk.x = p0; pk.y = p1;
    *(v2u*)(Aus + ((((half * 4 + m) * 9 + 0) * 32 + lp) * 8 + el)) = pk;
  }

  // per-lane parameters
  float om, c0;
  {
    const int h = 16 * wave + lo;
    om = omega[h];
    const float t2 = DT_C * om;
    c0 = (-1.0f + sqrtf(1.0f - t2 * t2)) / DT_C - boff[h];   // p_omega - b_offset
  }
  const float al0      = expf(-DT_C / tau[lane & 15]);       // wave-0 consumer
  const float one_m_al = 1.0f - al0;

  const v8f kZero = {0.f,0.f,0.f,0.f,0.f,0.f,0.f,0.f};
  v8f uS[4], vS[4], qS[4], acc[4];
  #pragma unroll
  for (int m = 0; m < 4; ++m) { uS[m] = kZero; vS[m] = kZero; qS[m] = kZero; }
  float nloc = 0.0f;

  // z-writer constants (operand-layout scatter for this wave's 16 h-columns)
  const int zHalf = wave & 1;
  const int zKk   = 1 + (wave >> 1);
  const int zS    = (lo >> 3) & 1;
  const int zEl   = lo & 7;

  __syncthreads();

  #pragma unroll 1
  for (int t = 0; t <= TT; ++t) {
    // prefetch x_{t+1} (global only; safe before barrier)
    v4f xv = {0.f,0.f,0.f,0.f};
    const bool havex = (t + 1 < TT);
    if (havex) xv = *(const v4f*)(x + (size_t)(t + 1) * (BB * II) + tid * 4);

    // ---- hidden GEMM: I_t = [x_t | z_{t-1}] @ Wh^T (this wave's 16 cols) ----
    if (t < TT) {
      #pragma unroll
      for (int m = 0; m < 4; ++m) acc[m] = kZero;
      #pragma unroll
      for (int kk = 0; kk < 9; ++kk) {
        ABu bb, a0, a1, a2, a3;
        bb.f4[0] = *(const v4f*)&WhB[0][wave][kk][lane];
        bb.f4[1] = *(const v4f*)&WhB[1][wave][kk][lane];
        a0.f4[0] = *(const v4f*)&Aop[0][0][kk][lane];
        a0.f4[1] = *(const v4f*)&Aop[1][0][kk][lane];
        a1.f4[0] = *(const v4f*)&Aop[0][1][kk][lane];
        a1.f4[1] = *(const v4f*)&Aop[1][1][kk][lane];
        a2.f4[0] = *(const v4f*)&Aop[0][2][kk][lane];
        a2.f4[1] = *(const v4f*)&Aop[1][2][kk][lane];
        a3.f4[0] = *(const v4f*)&Aop[0][3][kk][lane];
        a3.f4[1] = *(const v4f*)&Aop[1][3][kk][lane];
        acc[0] = __builtin_amdgcn_wmma_f32_16x16x32_bf16(false, a0.v, false, bb.v, (short)0, acc[0], false, false);
        acc[1] = __builtin_amdgcn_wmma_f32_16x16x32_bf16(false, a1.v, false, bb.v, (short)0, acc[1], false, false);
        acc[2] = __builtin_amdgcn_wmma_f32_16x16x32_bf16(false, a2.v, false, bb.v, (short)0, acc[2], false, false);
        acc[3] = __builtin_amdgcn_wmma_f32_16x16x32_bf16(false, a3.v, false, bb.v, (short)0, acc[3], false, false);
      }
    }

    // ---- output partial GEMM: z_{t-1} @ Wo^T (this wave: 1 k-block, 2 M-tiles) ----
    {
      ABu bb, a0, a1;
      bb.f4[0] = *(const v4f*)&WoB[0][kkO][lane];
      bb.f4[1] = *(const v4f*)&WoB[1][kkO][lane];
      a0.f4[0] = *(const v4f*)&Aop[0][mA][kkO + 1][lane];
      a0.f4[1] = *(const v4f*)&Aop[1][mA][kkO + 1][lane];
      a1.f4[0] = *(const v4f*)&Aop[0][mA + 1][kkO + 1][lane];
      a1.f4[1] = *(const v4f*)&Aop[1][mA + 1][kkO + 1][lane];
      v8f o0 = __builtin_amdgcn_wmma_f32_16x16x32_bf16(false, a0.v, false, bb.v, (short)0, kZero, false, false);
      v8f o1 = __builtin_amdgcn_wmma_f32_16x16x32_bf16(false, a1.v, false, bb.v, (short)0, kZero, false, false);
      #pragma unroll
      for (int r = 0; r < 8; ++r) {
        atomicAdd(&outpart[(16 * mA + 8 * hi + r) * OO + lo], o0[r]);
        atomicAdd(&outpart[(16 * (mA + 1) + 8 * hi + r) * OO + lo], o1[r]);
      }
    }

    __syncthreads();   // all A reads + outpart atomics complete

    // ---- state update + next-step staging ----
    if (t < TT) {
      #pragma unroll
      for (int m = 0; m < 4; ++m) {
        #pragma unroll
        for (int r = 0; r < 8; ++r) {
          const float It = acc[m][r];
          const float uo = uS[m][r], vo = vS[m][r], qo = qS[m][r];
          const float bb = c0 - qo;
          const float un = uo + DT_C * (bb * uo - om * vo + It);
          const float vn = vo + DT_C * (om * uo + bb * vo);
          const float zz = (un - 1.0f > 0.0f) ? 1.0f : 0.0f;
          uS[m][r] = un; vS[m][r] = vn; qS[m][r] = GAMMA_C * qo + zz;
          nloc += zz;
          const int lp = r + 8 * hi + 16 * zS;   // A-operand lane for row b
          Aus[(((zHalf * 4 + m) * 9 + zKk) * 32 + lp) * 8 + zEl] =
              (zz != 0.0f) ? (unsigned short)0x3F80 : (unsigned short)0;
        }
      }
      if (havex) {
        const int flat = tid * 4;
        const int b = flat >> 5, kb = flat & 31;
        const int m = b >> 4, half = (kb >> 4) & 1, s = (kb >> 3) & 1, el = kb & 7;
        const int lp = (b & 15) + 16 * s;
        unsigned int p0 = (unsigned int)f2bf(xv.x) | ((unsigned int)f2bf(xv.y) << 16);
        unsigned int p1 = (unsigned int)f2bf(xv.z) | ((unsigned int)f2bf(xv.w) << 16);
        v2u pk; pk.x = p0; pk.y = p1;
        *(v2u*)(Aus + ((((half * 4 + m) * 9 + 0) * 32 + lp) * 8 + el)) = pk;
      }
    }
    // wave 0: consume reduction -> LI EMA -> stream outputs[t-1]; re-zero outpart
    if (wave == 0 && t >= 1) {
      #pragma unroll 4
      for (int j = 0; j < 32; ++j) {
        const int e = lane + 32 * j;              // e & 15 == lane & 15 -> al0 matches o
        const float s  = outpart[e];
        const float nv = al0 * oubuf[e] + one_m_al * s;
        oubuf[e] = nv;
        out[(size_t)(t - 1) * (BB * OO) + e] = nv;
        outpart[e] = 0.0f;
      }
    }

    __syncthreads();   // z_t / x_{t+1} / zeroed outpart visible before next GEMM
  }

  // ---- final state outputs ----
  const size_t OFF_Z  = (size_t)TT * BB * OO;
  const size_t OFF_U  = OFF_Z + (size_t)BB * HH;
  const size_t OFF_OU = OFF_U + (size_t)BB * HH;
  const size_t OFF_NS = OFF_OU + (size_t)BB * OO;

  for (int idx = tid; idx < BB * HH; idx += NTH) {
    const int b = idx >> 8, h = idx & 255;
    const int kk = 1 + (h >> 5), kb = h & 31;
    const int half = (kb >> 4) & 1, s = (kb >> 3) & 1, el = kb & 7;
    const int lp = (b & 15) + 16 * s, m = b >> 4;
    out[OFF_Z + idx] = bf2f(Aus[(((half * 4 + m) * 9 + kk) * 32 + lp) * 8 + el]);
  }
  {
    const int h = 16 * wave + lo;
    #pragma unroll
    for (int m = 0; m < 4; ++m) {
      #pragma unroll
      for (int r = 0; r < 8; ++r) {
        const int b = 16 * m + 8 * hi + r;
        out[OFF_U + (size_t)b * HH + h] = uS[m][r];
      }
    }
  }
  for (int idx = tid; idx < BB * OO; idx += NTH) out[OFF_OU + idx] = oubuf[idx];
  atomicAdd(&ns_slot, nloc);
  __syncthreads();
  if (tid == 0) out[OFF_NS] = ns_slot;
}

extern "C" void kernel_launch(void* const* d_in, const int* in_sizes, int n_in,
                              void* d_out, int out_size, void* d_ws, size_t ws_size,
                              hipStream_t stream) {
  (void)in_sizes; (void)n_in; (void)d_ws; (void)ws_size; (void)out_size;
  const float* x     = (const float*)d_in[0];
  const float* Wh    = (const float*)d_in[1];
  const float* omega = (const float*)d_in[2];
  const float* boff  = (const float*)d_in[3];
  const float* Wo    = (const float*)d_in[4];
  const float* tau   = (const float*)d_in[5];
  float* out = (float*)d_out;
  brf_scan_kernel<<<dim3(1), dim3(NTH), 0, stream>>>(x, Wh, omega, boff, Wo, tau, out);
}
